// CustomAggregationLayerSimple_64364379897856
// MI455X (gfx1250) — matrix operand order repaired and checked by reading, compiled, not run on hardware
//
#include <hip/hip_runtime.h>

// ---------------------------------------------------------------------------
// Fused GraphSAGE layer for MI455X (gfx1250, wave32, WMMA):
//   out = relu((features + mean(neigh)) @ W + b)
// Memory-bound: 922MB total @ 23.3 TB/s => ~40us floor. GEMM (6.55 GFLOP)
// via v_wmma_f32_16x16x32_f16 (f32 accumulate) so matrix math stays off the
// critical path. This revision trades B-register footprint (128->64 VGPRs)
// for occupancy: 512 threads / 16 waves, one 16-col N-tile per wave.
// ---------------------------------------------------------------------------

typedef _Float16 v16h __attribute__((ext_vector_type(16)));
typedef _Float16 v8h  __attribute__((ext_vector_type(8)));
typedef _Float16 v4h  __attribute__((ext_vector_type(4)));
typedef float    v8f  __attribute__((ext_vector_type(8)));
typedef float    v4f  __attribute__((ext_vector_type(4)));

#define N_NODES 50000
#define D       256
#define NEI     16
#define ROWS    96            // node rows per block (6 M-tiles of 16)
#define THREADS 512           // 16 waves; wave w owns output cols [16w, 16w+16)
#define LDS_K   (D + 8)       // 264 halves -> 528B row stride (16B multiple, bank de-pad)

__global__ __launch_bounds__(THREADS)
void sage_fused_kernel(const float* __restrict__ feat,
                       const float* __restrict__ emb,
                       const float* __restrict__ wgt,   // [D][D] row-major (k, n)
                       const float* __restrict__ bias,
                       float* __restrict__ out)
{
    // x-tile staged as f16 in WMMA-A row-major layout: 96 x 264 halves = 50.7KB
    __shared__ __align__(16) _Float16 xT[ROWS * LDS_K];

    const int tid  = threadIdx.x;
    const int lane = tid & 31;
    const int wave = tid >> 5;          // 0..15
    const int row0 = blockIdx.x * ROWS;

    // ---------------- Phase 1: mean-aggregate + add self, f32 -> f16 -> LDS ----
    // 96 rows x 64 float4-chunks; 512 threads sweep 8 rows/iter, fully coalesced
    // (32 lanes x 16B = 512B contiguous per wave). Non-temporal: read-once stream.
    // Two accumulators break the 16-deep add chain.
    #pragma unroll 2
    for (int it = 0; it < ROWS / 8; ++it) {
        const int r  = it * 8 + (tid >> 6);   // 0..95  (uniform per wave)
        const int c4 = tid & 63;              // float4 chunk 0..63
        const int gr = row0 + r;
        v4f x = {0.f, 0.f, 0.f, 0.f};
        if (gr < N_NODES) {
            const v4f f = __builtin_nontemporal_load(
                (const v4f*)(feat + (size_t)gr * D) + c4);
            const float* ebase = emb + (size_t)gr * NEI * D + c4 * 4;
            v4f acc0 = {0.f, 0.f, 0.f, 0.f};
            v4f acc1 = {0.f, 0.f, 0.f, 0.f};
            #pragma unroll
            for (int j = 0; j < NEI; j += 2) {
                acc0 += __builtin_nontemporal_load((const v4f*)(ebase + (size_t)j * D));
                acc1 += __builtin_nontemporal_load((const v4f*)(ebase + (size_t)(j + 1) * D));
            }
            x = f + (acc0 + acc1) * (1.0f / NEI);
        }
        v4h h = __builtin_convertvector(x, v4h);
        *(v4h*)(&xT[r * LDS_K + c4 * 4]) = h;   // 8B aligned (stride 528, off 8*c4)
    }

    // ---------------- Phase 2: weight B-tile -> registers (f16) ----------------
    // Wave handles ONE N-tile of 16 cols. B 32x16 f16 lane layout:
    //   lanes 0-15 : elems 0-7 hold K=0..7,  elems 8-15 K=16..23 for col N=lane
    //   lanes 16-31: elems 0-7 hold K=8..15, elems 8-15 K=24..31 for col N=lane-16
    // 8 x v16h = 64 VGPRs/lane. Weights are L2-hot (256KB << 192MB L2).
    const int nsub  = lane & 15;
    const int khalf = (lane & 16) ? 16 : 0;
    const int ncol  = wave * 16 + nsub;
    v16h B[8];
    #pragma unroll
    for (int ks = 0; ks < 8; ++ks) {
        const int kbase = ks * 32 + (khalf >> 1);
        v16h b;
        #pragma unroll
        for (int i = 0; i < 16; ++i)
            b[i] = (_Float16)wgt[(kbase + i + (i & 8)) * D + ncol];
        B[ks] = b;
    }

    __syncthreads();

    // ---------------- Phase 3: WMMA GEMM + bias + relu -------------------------
    const float bv  = bias[ncol];
    const int  aoff = (lane & 16) ? 8 : 0;  // A lane-group K offset per ISA layout

    #pragma unroll
    for (int mt = 0; mt < ROWS / 16; ++mt) {
        v8f c = {};
        const _Float16* arow = &xT[(mt * 16 + nsub) * LDS_K];
        #pragma unroll
        for (int ks = 0; ks < 8; ++ks) {
            const _Float16* p = arow + ks * 32 + aoff;
            v8h lo = *(const v8h*)(p);         // ds_load_b128, 16B aligned
            v8h hi = *(const v8h*)(p + 16);    // ds_load_b128
            v16h a = __builtin_shufflevector(lo, hi,
                        0,1,2,3,4,5,6,7,8,9,10,11,12,13,14,15);
            c = __builtin_amdgcn_wmma_f32_16x16x32_f16(
                    false, a, false, B[ks], (short)0, c, false, false);
        }
        // C/D layout: VGPR v -> M = v (lanes 0-15) or v+8 (lanes 16-31); N = lane%16
        const int mbase = row0 + mt * 16 + ((lane & 16) ? 8 : 0);
        #pragma unroll
        for (int v = 0; v < 8; ++v) {
            const int gr = mbase + v;
            if (gr < N_NODES) {
                float o = c[v] + bv;
                o = o > 0.f ? o : 0.f;
                __builtin_nontemporal_store(o, out + (size_t)gr * D + ncol);
            }
        }
    }
}

extern "C" void kernel_launch(void* const* d_in, const int* in_sizes, int n_in,
                              void* d_out, int out_size, void* d_ws, size_t ws_size,
                              hipStream_t stream) {
    const float* feat = (const float*)d_in[0];   // [50000, 256]
    const float* emb  = (const float*)d_in[1];   // [50000, 16, 256]
    const float* wgt  = (const float*)d_in[2];   // [256, 256]
    const float* bias = (const float*)d_in[3];   // [256]
    float* out = (float*)d_out;                  // [50000, 256]

    const int grid = (N_NODES + ROWS - 1) / ROWS;   // 521 blocks
    sage_fused_kernel<<<grid, THREADS, 0, stream>>>(feat, emb, wgt, bias, out);
}
